// EpisodicMemoryModule_83588653514883
// MI455X (gfx1250) — compile-verified, matrix operand order checked by reading, not run on hardware
//
#include <hip/hip_runtime.h>
#include <hip/hip_bf16.h>
#include <math.h>

// ---------------------------------------------------------------------------
// Problem constants: B=128, N=1024, D=256, STEPS=3
// ---------------------------------------------------------------------------
#define BB   128
#define NN   1024
#define DD   256
#define ZD   1024          // 4*D
#define MTOT (BB * NN)

typedef float  v8f  __attribute__((ext_vector_type(8)));
typedef __bf16 v16bf __attribute__((ext_vector_type(16)));
typedef __bf16 v8bf  __attribute__((ext_vector_type(8)));
typedef __bf16 v4bf  __attribute__((ext_vector_type(4)));
typedef unsigned int u32x4 __attribute__((ext_vector_type(4)));
typedef int i32x4 __attribute__((ext_vector_type(4)));
typedef int i32x8 __attribute__((ext_vector_type(8)));

#define WMMA_BF16(a, b, c) \
  __builtin_amdgcn_wmma_f32_16x16x32_bf16(false, (a), false, (b), (short)0, (c), false, false)

__device__ __forceinline__ v16bf frag_combine(v8bf lo, v8bf hi) {
  v16bf r;
#pragma unroll
  for (int e = 0; e < 8; ++e) { r[e] = lo[e]; r[8 + e] = hi[e]; }
  return r;
}

// A fragment 16x32 (ISA 7.12.2): lane m=lane&15, K-half (lane>>4)*8.
__device__ __forceinline__ v16bf load_a16x32_lds(const __bf16* base, int ld) {
  int lane = threadIdx.x & 31;
  const __bf16* p = base + (size_t)(lane & 15) * ld + ((lane >> 4) << 3);
  v8bf lo = *(const v8bf*)p;
  v8bf hi = *(const v8bf*)(p + 16);
  return frag_combine(lo, hi);
}

// B fragment from fragment-ordered storage: 32 lanes x 16 contiguous elements.
__device__ __forceinline__ v16bf load_b_frag(const __bf16* chunk512) {
  int lane = threadIdx.x & 31;
  const __bf16* p = chunk512 + lane * 16;
  v8bf lo = *(const v8bf*)p;
  v8bf hi = *(const v8bf*)(p + 8);
  return frag_combine(lo, hi);
}

// ---------------------------------------------------------------------------
// Tensor Data Mover: 1-D bulk global->LDS DMA (ISA ch.8 D# descriptor).
// data_size = 8B elements; tensor_dim0 = tile_dim0 = stride = nelem8.
// Issue from one wave; completion via TENSORcnt.
// ---------------------------------------------------------------------------
__device__ __forceinline__ unsigned lds_offset(const void* p) {
  return (unsigned)(unsigned long long)(__attribute__((address_space(3))) const char*)p;
}

__device__ __forceinline__ void tdm_load_1d(unsigned lds_addr, const void* gsrc,
                                            unsigned nelem8) {
  unsigned long long ga = (unsigned long long)gsrc;
  u32x4 g0;
  g0[0] = 1u;                                        // count = 1 valid descriptor
  g0[1] = lds_addr;                                  // LDS byte address
  g0[2] = (unsigned)(ga & 0xFFFFFFFFu);              // global_addr[31:0]
  g0[3] = (unsigned)((ga >> 32) & 0x01FFFFFFu) | (2u << 30);  // addr[56:32] | type=2
  i32x8 g1;
  g1[0] = (int)(3u << 16);                           // data_size = 3 (8 bytes)
  g1[1] = (int)((nelem8 & 0xFFFFu) << 16);           // tensor_dim0[15:0]  (bits 63:48)
  g1[2] = (int)((nelem8 >> 16) & 0xFFFFu);           // tensor_dim0[31:16] (bits 79:64)
  g1[3] = (int)((nelem8 & 0xFFFFu) << 16);           // tile_dim0 (bits 127:112)
  g1[4] = 0;                                         // tile_dim1/2 unused
  g1[5] = (int)nelem8;                               // tensor_dim0_stride[31:0]
  g1[6] = 0;
  g1[7] = 0;
#if __clang_major__ >= 23
  __builtin_amdgcn_tensor_load_to_lds(g0, g1, (i32x4){0, 0, 0, 0}, (i32x4){0, 0, 0, 0},
                                      (i32x8){0, 0, 0, 0, 0, 0, 0, 0}, 0);
#else
  __builtin_amdgcn_tensor_load_to_lds(g0, g1, (i32x4){0, 0, 0, 0}, (i32x4){0, 0, 0, 0}, 0);
#endif
}

// ---------------------------------------------------------------------------
// fp32 -> bf16 (grid-stride) for the A-side facts matrix
// ---------------------------------------------------------------------------
__global__ void cvt_f32_bf16(const float* __restrict__ src, __bf16* __restrict__ dst,
                             unsigned long long n) {
  unsigned long long i = (unsigned long long)blockIdx.x * blockDim.x + threadIdx.x;
  unsigned long long stride = (unsigned long long)gridDim.x * blockDim.x;
  for (; i < n; i += stride) dst[i] = (__bf16)src[i];
}

// ---------------------------------------------------------------------------
// One-time swizzle of a B matrix (K x N logical) into WMMA fragment order:
//   dst[((nt*(K/32) + kt)*32 + lane)*16 + e],  lane = (n&15) + 16*((k>>3)&1)
// transpose=1: B[k][n] = src[n*K + k]   (W1: h1 = z @ W1^T)
// ---------------------------------------------------------------------------
__global__ void swizzle_b(const float* __restrict__ src, __bf16* __restrict__ dst,
                          int K, int N, int transpose) {
  int total = K * N;
  int ktiles = K >> 5;
  for (int i = blockIdx.x * blockDim.x + threadIdx.x; i < total;
       i += gridDim.x * blockDim.x) {
    int e    = i & 15;
    int lane = (i >> 4) & 31;
    int tile = i >> 9;
    int nt = tile / ktiles, kt = tile - nt * ktiles;
    int nc = lane & 15, half = lane >> 4;
    int kk = (e & 7) + (half << 3) + ((e >> 3) << 4);
    int k = (kt << 5) + kk;
    int n = (nt << 4) + nc;
    float v = transpose ? src[(size_t)n * K + k] : src[(size_t)k * N + n];
    dst[i] = (__bf16)v;
  }
}

// ---------------------------------------------------------------------------
// GEMM + bias: C[M,N] = A[M,K]bf16 @ Bfrag[K,N]bf16 + bias[N]  (fp32 out)
// BLK_M=128, BLK_N=128; 256 threads; wave -> 16 M-rows, 8 N-tiles.
// ---------------------------------------------------------------------------
__global__ void gemm_bf16_bias(const __bf16* __restrict__ A, const __bf16* __restrict__ BF,
                               const float* __restrict__ bias, float* __restrict__ C,
                               int M, int N, int K) {
  __shared__ __align__(16) __bf16 As[128][40];
  __shared__ __align__(16) __bf16 BsF[8][512];
  const int wave = threadIdx.x >> 5;
  const int lane = threadIdx.x & 31;
  const int m0 = blockIdx.x * 128;
  const int n0 = blockIdx.y * 128;
  const int n0t = blockIdx.y * 8;
  const int ktiles = K >> 5;

  v8f acc[8];
#pragma unroll
  for (int t = 0; t < 8; ++t) acc[t] = (v8f){0.f,0.f,0.f,0.f,0.f,0.f,0.f,0.f};

  for (int kt = 0; kt < ktiles; ++kt) {
    for (int i = threadIdx.x; i < 512; i += 256) {
      int r = i >> 2, c8 = (i & 3) << 3;
      *(uint4*)&As[r][c8] = *(const uint4*)&A[(size_t)(m0 + r) * K + (kt << 5) + c8];
    }
    for (int i = threadIdx.x; i < 512; i += 256) {
      int nt = i >> 6, u = (i & 63) << 3;
      *(uint4*)&BsF[nt][u] =
          *(const uint4*)&BF[((size_t)(n0t + nt) * ktiles + kt) * 512 + u];
    }
    __syncthreads();
    // load all fragments first, then run the WMMA chain (pipelined waits)
    v16bf a = load_a16x32_lds(&As[wave * 16][0], 40);
    v16bf bf[8];
#pragma unroll
    for (int nt = 0; nt < 8; ++nt) bf[nt] = load_b_frag(&BsF[nt][0]);
#pragma unroll
    for (int nt = 0; nt < 8; ++nt) acc[nt] = WMMA_BF16(a, bf[nt], acc[nt]);
    __syncthreads();
  }
  const int mbase = wave * 16 + ((lane >> 4) << 3);
  const int ncol  = lane & 15;
#pragma unroll
  for (int nt = 0; nt < 8; ++nt) {
    int n = n0 + nt * 16 + ncol;
    float bv = bias[n];
#pragma unroll
    for (int v = 0; v < 8; ++v)
      C[(size_t)(m0 + mbase + v) * N + n] = acc[nt][v] + bv;
  }
}

// ---------------------------------------------------------------------------
// Fused attention score: z built on the fly, h1 = tanh(z@W1^T + b1),
// score = h1.W2 + b2.  grid (N/128, B), 256 threads.
// ---------------------------------------------------------------------------
__global__ void attention_kernel(const float* __restrict__ facts,
                                 const float* __restrict__ q,
                                 const float* __restrict__ mem,
                                 const __bf16* __restrict__ W1F,
                                 const float* __restrict__ b1,
                                 const float* __restrict__ W2,
                                 const float* __restrict__ b2,
                                 float* __restrict__ scores) {
  __shared__ float qs[DD];
  __shared__ float ms[DD];
  __shared__ __align__(16) __bf16 Zs[128][40];
  __shared__ __align__(16) __bf16 BsF[16][512];
  const int b  = blockIdx.y;
  const int n0 = blockIdx.x * 128;
  const int wave = threadIdx.x >> 5;
  const int lane = threadIdx.x & 31;
  const int ktiles = ZD >> 5;   // 32

  qs[threadIdx.x] = q[(size_t)b * DD + threadIdx.x];
  ms[threadIdx.x] = mem[(size_t)b * DD + threadIdx.x];

  v8f acc[16];
#pragma unroll
  for (int t = 0; t < 16; ++t) acc[t] = (v8f){0.f,0.f,0.f,0.f,0.f,0.f,0.f,0.f};
  __syncthreads();

  for (int kt = 0; kt < ktiles; ++kt) {
    const int k0 = kt << 5;
    const int zone = k0 >> 8;
    const int dk0  = k0 & 255;
    for (int i = threadIdx.x; i < 1024; i += 256) {
      int r = i >> 3, c4 = (i & 7) << 2;
      float4 f = *(const float4*)&facts[((size_t)b * NN + n0 + r) * DD + dk0 + c4];
      v4bf z;
      const float* ov = (zone == 0 || zone == 2) ? qs : ms;
      float o0 = ov[dk0 + c4], o1 = ov[dk0 + c4 + 1], o2 = ov[dk0 + c4 + 2], o3 = ov[dk0 + c4 + 3];
      if (zone < 2) {
        z[0] = (__bf16)(f.x * o0); z[1] = (__bf16)(f.y * o1);
        z[2] = (__bf16)(f.z * o2); z[3] = (__bf16)(f.w * o3);
      } else {
        z[0] = (__bf16)fabsf(f.x - o0); z[1] = (__bf16)fabsf(f.y - o1);
        z[2] = (__bf16)fabsf(f.z - o2); z[3] = (__bf16)fabsf(f.w - o3);
      }
      *(v4bf*)&Zs[r][c4] = z;
    }
    for (int i = threadIdx.x; i < 1024; i += 256) {
      int nt = i >> 6, u = (i & 63) << 3;
      *(uint4*)&BsF[nt][u] = *(const uint4*)&W1F[((size_t)nt * ktiles + kt) * 512 + u];
    }
    __syncthreads();
    v16bf a = load_a16x32_lds(&Zs[wave * 16][0], 40);
#pragma unroll
    for (int nt = 0; nt < 16; ++nt) {
      v16bf bfr = load_b_frag(&BsF[nt][0]);
      acc[nt] = WMMA_BF16(a, bfr, acc[nt]);
    }
    __syncthreads();
  }

  const int ncol = lane & 15;
  float part[8];
#pragma unroll
  for (int v = 0; v < 8; ++v) part[v] = 0.f;
#pragma unroll
  for (int nt = 0; nt < 16; ++nt) {
    int d = nt * 16 + ncol;
    float w2v = W2[d], b1v = b1[d];
#pragma unroll
    for (int v = 0; v < 8; ++v) part[v] += tanhf(acc[nt][v] + b1v) * w2v;
  }
#pragma unroll
  for (int off = 1; off < 16; off <<= 1) {
#pragma unroll
    for (int v = 0; v < 8; ++v) part[v] += __shfl_xor(part[v], off, 32);
  }
  if ((lane & 15) == 0) {
    int mrow = wave * 16 + ((lane >> 4) << 3);
#pragma unroll
    for (int v = 0; v < 8; ++v)
      scores[(size_t)b * NN + n0 + mrow + v] = part[v] + b2[0];
  }
}

// ---------------------------------------------------------------------------
// Softmax over N=1024 per batch (one WG of 256 per b)
// ---------------------------------------------------------------------------
__global__ void softmax_kernel(const float* __restrict__ scores, float* __restrict__ gates) {
  __shared__ float red[256];
  const int b = blockIdx.x, t = threadIdx.x;
  const float* s = scores + (size_t)b * NN;
  float v[4];
  float mx = -INFINITY;
#pragma unroll
  for (int i = 0; i < 4; ++i) { v[i] = s[t + i * 256]; mx = fmaxf(mx, v[i]); }
  red[t] = mx; __syncthreads();
  for (int o = 128; o > 0; o >>= 1) { if (t < o) red[t] = fmaxf(red[t], red[t + o]); __syncthreads(); }
  mx = red[0]; __syncthreads();
  float sum = 0.f;
#pragma unroll
  for (int i = 0; i < 4; ++i) { v[i] = expf(v[i] - mx); sum += v[i]; }
  red[t] = sum; __syncthreads();
  for (int o = 128; o > 0; o >>= 1) { if (t < o) red[t] += red[t + o]; __syncthreads(); }
  float inv = 1.f / red[0];
#pragma unroll
  for (int i = 0; i < 4; ++i) gates[(size_t)b * NN + t + i * 256] = v[i] * inv;
}

// ---------------------------------------------------------------------------
// Persistent GRU scan: 8 WGs x 256 thr; WG owns 16 batch rows for 1024 steps.
// Ur/Uh (fragment-ordered, 256KB) pulled into LDS by the Tensor Data Mover;
// per step two 16x256x256 WMMA GEMMs + fused gate update.
// ---------------------------------------------------------------------------
__global__ void gru_scan_kernel(const float* __restrict__ FWr, const float* __restrict__ FWh,
                                const __bf16* __restrict__ UrF, const __bf16* __restrict__ UhF,
                                const float* __restrict__ gates, float* __restrict__ e_out) {
  __shared__ __align__(16) __bf16 Us[2 * 65536];   // 262,144 B (frag-ordered Ur,Uh)
  __shared__ __align__(16) __bf16 hbf[16][264];
  __shared__ __align__(16) float  hs[16][DD];
  __shared__ __align__(16) float  rub[2][16][DD];
  const int b0   = blockIdx.x * 16;
  const int wave = threadIdx.x >> 5;
  const int lane = threadIdx.x & 31;

  // TDM bulk DMA: 2 x 128KB global -> LDS, issued by wave 0 only
  if (threadIdx.x == 0) {
    tdm_load_1d(lds_offset(&Us[0]),     UrF, 16384);   // 16384 x 8B = 128KB
    tdm_load_1d(lds_offset(&Us[65536]), UhF, 16384);
  }
  for (int i = threadIdx.x; i < 16 * DD / 4; i += 256) {
    int r = i >> 6, d4 = (i & 63) << 2;
    *(float4*)&hs[r][d4] = (float4){0.f, 0.f, 0.f, 0.f};
    *(v4bf*)&hbf[r][d4] = (v4bf){(__bf16)0.f, (__bf16)0.f, (__bf16)0.f, (__bf16)0.f};
  }
  if (threadIdx.x == 0) __builtin_amdgcn_s_wait_tensorcnt((short)0);
  __syncthreads();

  for (int n = 0; n < NN; ++n) {
    // prefetch this step's FW rows while the GEMM runs
    {
      size_t base = ((size_t)b0 * NN + n) * DD;
      int r = threadIdx.x >> 4;
      int c = (threadIdx.x & 15) * 16;
      size_t idx = base + (size_t)r * NN * DD + c;
      __builtin_prefetch(&FWr[idx], 0, 3);
      __builtin_prefetch(&FWh[idx], 0, 3);
    }
    // h(16x256) @ {Ur,Uh}: 32 output tile-cols, 4 per wave
    v8f acc[4];
#pragma unroll
    for (int t = 0; t < 4; ++t) acc[t] = (v8f){0.f,0.f,0.f,0.f,0.f,0.f,0.f,0.f};
#pragma unroll
    for (int kt = 0; kt < 8; ++kt) {
      v16bf a = load_a16x32_lds(&hbf[0][kt << 5], 264);
      v16bf bfr[4];
#pragma unroll
      for (int t = 0; t < 4; ++t) {
        int tc = wave * 4 + t;
        bfr[t] = load_b_frag(&Us[(tc >> 4) * 65536 + ((tc & 15) * 8 + kt) * 512]);
      }
#pragma unroll
      for (int t = 0; t < 4; ++t) acc[t] = WMMA_BF16(a, bfr[t], acc[t]);
    }
    const int mbase = (lane >> 4) << 3;
    const int ncol  = lane & 15;
#pragma unroll
    for (int t = 0; t < 4; ++t) {
      int tc = wave * 4 + t;
      int which = tc >> 4;
      int nnc = (tc & 15) * 16 + ncol;
#pragma unroll
      for (int v = 0; v < 8; ++v) rub[which][mbase + v][nnc] = acc[t][v];
    }
    __syncthreads();
    // fused gate update
    for (int i = threadIdx.x; i < 16 * DD / 4; i += 256) {
      int r = i >> 6, d4 = (i & 63) << 2;
      size_t idx = ((size_t)(b0 + r) * NN + n) * DD + d4;
      float4 fr = *(const float4*)&FWr[idx];
      float4 fh = *(const float4*)&FWh[idx];
      float4 ur = *(const float4*)&rub[0][r][d4];
      float4 uh = *(const float4*)&rub[1][r][d4];
      float g = gates[(size_t)(b0 + r) * NN + n];
      float4 ho = *(const float4*)&hs[r][d4];
      float4 hn;
      {
        float rr0 = 1.f / (1.f + expf(-(fr.x + ur.x)));
        float rr1 = 1.f / (1.f + expf(-(fr.y + ur.y)));
        float rr2 = 1.f / (1.f + expf(-(fr.z + ur.z)));
        float rr3 = 1.f / (1.f + expf(-(fr.w + ur.w)));
        hn.x = g * tanhf(fh.x + rr0 * uh.x) + (1.f - g) * ho.x;
        hn.y = g * tanhf(fh.y + rr1 * uh.y) + (1.f - g) * ho.y;
        hn.z = g * tanhf(fh.z + rr2 * uh.z) + (1.f - g) * ho.z;
        hn.w = g * tanhf(fh.w + rr3 * uh.w) + (1.f - g) * ho.w;
      }
      *(float4*)&hs[r][d4] = hn;
      v4bf hb;
      hb[0] = (__bf16)hn.x; hb[1] = (__bf16)hn.y; hb[2] = (__bf16)hn.z; hb[3] = (__bf16)hn.w;
      *(v4bf*)&hbf[r][d4] = hb;
    }
    __syncthreads();
  }
  for (int i = threadIdx.x; i < 16 * DD / 4; i += 256) {
    int r = i >> 6, d4 = (i & 63) << 2;
    *(float4*)&e_out[(size_t)(b0 + r) * DD + d4] = *(const float4*)&hs[r][d4];
  }
}

// ---------------------------------------------------------------------------
// memory = relu([mem, e, q] @ Wm^T + bm)
// ---------------------------------------------------------------------------
__global__ void memory_update_kernel(const float* __restrict__ mem_in,
                                     const float* __restrict__ e,
                                     const float* __restrict__ q,
                                     const float* __restrict__ Wm,
                                     const float* __restrict__ bm,
                                     float* __restrict__ mem_out) {
  const int b = blockIdx.x, d = threadIdx.x;
  const float* w = &Wm[(size_t)d * (3 * DD)];
  float s = bm[d];
  for (int k = 0; k < DD; ++k) s += mem_in[(size_t)b * DD + k] * w[k];
  for (int k = 0; k < DD; ++k) s += e[(size_t)b * DD + k] * w[DD + k];
  for (int k = 0; k < DD; ++k) s += q[(size_t)b * DD + k] * w[2 * DD + k];
  mem_out[(size_t)b * DD + d] = fmaxf(s, 0.f);
}

// ---------------------------------------------------------------------------
// Host orchestration
// ---------------------------------------------------------------------------
extern "C" void kernel_launch(void* const* d_in, const int* in_sizes, int n_in,
                              void* d_out, int out_size, void* d_ws, size_t ws_size,
                              hipStream_t stream) {
  (void)in_sizes; (void)n_in; (void)out_size; (void)ws_size;
  const float* facts = (const float*)d_in[0];
  const float* question = (const float*)d_in[1];
  const float* W1 = (const float*)d_in[2];
  const float* b1 = (const float*)d_in[3];
  const float* W2 = (const float*)d_in[4];
  const float* b2 = (const float*)d_in[5];
  const float* Wr = (const float*)d_in[6];
  const float* Ur = (const float*)d_in[7];
  const float* br = (const float*)d_in[8];
  const float* Wh = (const float*)d_in[9];
  const float* Uh = (const float*)d_in[10];
  const float* bh = (const float*)d_in[11];
  const float* Wm = (const float*)d_in[12];
  const float* bm = (const float*)d_in[13];

  char* ws = (char*)d_ws;
  const size_t BND = (size_t)BB * NN * DD;
  float*  FWr    = (float*)(ws);
  float*  FWh    = (float*)(ws + BND * 4);
  __bf16* factsb = (__bf16*)(ws + BND * 8);
  size_t  off    = BND * 8 + BND * 2;
  __bf16* W1F    = (__bf16*)(ws + off); off += (size_t)DD * ZD * 2;
  __bf16* WrF    = (__bf16*)(ws + off); off += (size_t)DD * DD * 2;
  __bf16* WhF    = (__bf16*)(ws + off); off += (size_t)DD * DD * 2;
  __bf16* UrF    = (__bf16*)(ws + off); off += (size_t)DD * DD * 2;
  __bf16* UhF    = (__bf16*)(ws + off); off += (size_t)DD * DD * 2;
  float*  scores = (float*)(ws + off);  off += (size_t)BB * NN * 4;
  float*  gates  = (float*)(ws + off);  off += (size_t)BB * NN * 4;
  float*  memA   = (float*)(ws + off);  off += (size_t)BB * DD * 4;
  float*  memB   = (float*)(ws + off);  off += (size_t)BB * DD * 4;
  float*  ebuf   = (float*)(ws + off);

  cvt_f32_bf16<<<4096, 256, 0, stream>>>(facts, factsb, BND);
  swizzle_b<<<512, 256, 0, stream>>>(W1, W1F, ZD, DD, 1);
  swizzle_b<<<128, 256, 0, stream>>>(Wr, WrF, DD, DD, 0);
  swizzle_b<<<128, 256, 0, stream>>>(Wh, WhF, DD, DD, 0);
  swizzle_b<<<128, 256, 0, stream>>>(Ur, UrF, DD, DD, 0);
  swizzle_b<<<128, 256, 0, stream>>>(Uh, UhF, DD, DD, 0);

  dim3 gproj(MTOT / 128, DD / 128);
  gemm_bf16_bias<<<gproj, 256, 0, stream>>>(factsb, WrF, br, FWr, MTOT, DD, DD);
  gemm_bf16_bias<<<gproj, 256, 0, stream>>>(factsb, WhF, bh, FWh, MTOT, DD, DD);

  hipMemcpyAsync(memA, question, (size_t)BB * DD * 4, hipMemcpyDeviceToDevice, stream);

  float* cur = memA;
  float* nxt = memB;
  for (int step = 0; step < 3; ++step) {
    attention_kernel<<<dim3(NN / 128, BB), 256, 0, stream>>>(
        facts, question, cur, W1F, b1, W2, b2, scores);
    softmax_kernel<<<BB, 256, 0, stream>>>(scores, gates);
    gru_scan_kernel<<<BB / 16, 256, 0, stream>>>(FWr, FWh, UrF, UhF, gates, ebuf);
    float* dst = (step == 2) ? (float*)d_out : nxt;
    memory_update_kernel<<<BB, 256, 0, stream>>>(cur, ebuf, question, Wm, bm, dst);
    float* tmp = cur; cur = nxt; nxt = tmp;
  }
}
